// CTMCell_74835510165728
// MI455X (gfx1250) — matrix operand order repaired and checked by reading, not compiled, not run on hardware
//
#include <hip/hip_runtime.h>
#include <hip/hip_bf16.h>

// ---------------------------------------------------------------------------
// CTM cell for MI455X (gfx1250). All matmuls via v_wmma_f32_16x16x32_f16.
// Weights pre-packed per call into WMMA B-fragment layout so the GEMM hot loop
// does contiguous b128 loads instead of u16 gathers. st/at kept as (B,D,36)
// ring-extended buffers; iteration t appends column 32+t, outputs are cols 4..35.
// ---------------------------------------------------------------------------

#define B_     256
#define OBS_   256
#define DIN_   512
#define D_     1024
#define M_     32
#define HID_   16
#define NSY_   64
#define ITERS_ 4
#define SYNCH_ 2080
#define MEXT_  36          // M_ + ITERS_

typedef __attribute__((ext_vector_type(16))) _Float16 v16h;
typedef __attribute__((ext_vector_type(8)))  float    v8f;

union AFrag {              // one 16x32 f16 WMMA operand (8 VGPRs / lane)
  v16h     v;
  unsigned u[8];
  _Float16 h[16];
};

__device__ __forceinline__ float fast_sigmoid(float x) {
  // a * sigmoid(b) epilogues: v_rcp_f32 instead of IEEE div chains
  return __builtin_amdgcn_rcpf(1.0f + __expf(-x));
}

// ---------------------------------------------------------------- f32 -> f16
__global__ void k_cvt_f16(const float* __restrict__ in, _Float16* __restrict__ out, int n) {
  int i = blockIdx.x * blockDim.x + threadIdx.x;
  if (i < n) out[i] = (_Float16)in[i];
}

// ------------------------------------------- pack W (K x N2 f32) into WMMA
// fragment layout: Wp[((ks*NT2 + ct)*32 + lane)*16 + e], where e=2v+lo maps to
// k = ks*32 + 2v + 16*(lane>>4) + lo, col = ct*16 + (lane&15).
__global__ void k_pack_w(const float* __restrict__ W, _Float16* __restrict__ Wp, int K, int N2) {
  int i = blockIdx.x * blockDim.x + threadIdx.x;
  if (i >= K * N2) return;
  int e    = i & 15;
  int lane = (i >> 4) & 31;
  int rest = i >> 9;
  int nt2  = N2 >> 4;
  int ct   = rest % nt2;
  int ks   = rest / nt2;
  int v = e >> 1, lo = e & 1, half = lane >> 4;
  int k   = ks * 32 + 2 * v + 16 * half + lo;
  int col = ct * 16 + (lane & 15);
  Wp[i] = (_Float16)W[(size_t)k * N2 + col];
}

// ------------------- pack nlm1_w (M,2H,D f32) into per-d WMMA B fragments
// Wp[((d*2 + tile)*32 + lane)*16 + e]: m = 2v+16*(lane>>4)+lo, h = tile*16+(lane&15)
__global__ void k_pack_nlm(const float* __restrict__ w1, _Float16* __restrict__ Wp) {
  int i = blockIdx.x * blockDim.x + threadIdx.x;
  if (i >= D_ * 2 * 32 * 16) return;
  int e    = i & 15;
  int lane = (i >> 4) & 31;
  int tile = (i >> 9) & 1;
  int d    = i >> 10;
  int v = e >> 1, lo = e & 1, half = lane >> 4;
  int m = 2 * v + 16 * half + lo;
  int h = tile * 16 + (lane & 15);
  Wp[i] = (_Float16)w1[((size_t)m * 32 + h) * D_ + d];
}

// -------------------------------------------- init st/at ext with done-reset
__global__ void k_init_ext(const float* __restrict__ st_in, const float* __restrict__ at_in,
                           const float* __restrict__ st0,   const float* __restrict__ at0,
                           const unsigned char* __restrict__ dones,
                           float* __restrict__ st_ext, float* __restrict__ at_ext) {
  int i = blockIdx.x * blockDim.x + threadIdx.x;      // i = b*D + d
  if (i >= B_ * D_) return;
  int b = i / D_, d = i % D_;
  bool rst = dones[b] != 0;
  const float4* ssrc = (const float4*)(rst ? (st0 + (size_t)d * M_) : (st_in + (size_t)i * M_));
  const float4* asrc = (const float4*)(rst ? (at0 + (size_t)d * M_) : (at_in + (size_t)i * M_));
  float4* sdst = (float4*)(st_ext + (size_t)i * MEXT_);
  float4* adst = (float4*)(at_ext + (size_t)i * MEXT_);
#pragma unroll
  for (int q = 0; q < 8; ++q) { sdst[q] = ssrc[q]; adst[q] = asrc[q]; }
}

// ------------------------------------------------- GEMM + bias + GLU (WMMA)
// A: Mr x K (f16, ld=lda)  Wp: packed fragments of K x 2N weights  bias: 2N
// out: Mr x N f32, out[r,c] = (AW+b)[r,c] * sigmoid((AW+b)[r,c+N])
__global__ void __launch_bounds__(128)
k_gemm_glu(const _Float16* __restrict__ A, int lda,
           const _Float16* __restrict__ Wp,
           const float* __restrict__ bias,
           float* __restrict__ outg, int K, int N) {
  const int lane = threadIdx.x & 31;
  const int wave = blockIdx.x * (blockDim.x >> 5) + (threadIdx.x >> 5);
  const int NT   = N >> 4;          // glu-half col tiles
  const int NT2  = NT * 2;
  const int rt   = wave / NT;
  const int ct   = wave % NT;
  const int half = lane >> 4;
  const int col  = ct * 16 + (lane & 15);
  const _Float16* __restrict__ arow = A + (size_t)(rt * 16 + (lane & 15)) * lda;
  const size_t kstride = (size_t)NT2 * 512;              // halves per k-step
  const _Float16* __restrict__ bp0 = Wp + ((size_t)ct * 32 + lane) * 16;
  const _Float16* __restrict__ bp1 = Wp + ((size_t)(ct + NT) * 32 + lane) * 16;

  v8f acc0 = {}; v8f acc1 = {};
  const int KS = K >> 5;
  for (int ks = 0; ks < KS; ++ks) {
    AFrag a;
#pragma unroll
    for (int v = 0; v < 8; ++v) {
      // A fragment in this code's contiguous k order (matches packed B): VGPR v: K=2v,2v+1 (+16 lanes 16-31)
      int kA = 2 * v + 16 * half;
      a.u[v] = *(const unsigned*)(arow + ks * 32 + kA);
    }
    v16h b0 = *(const v16h*)(bp0 + ks * kstride);        // contiguous 32B (2x b128)
    v16h b1 = *(const v16h*)(bp1 + ks * kstride);
    if (ks + 1 < KS) {                                    // global_prefetch_b8
      __builtin_prefetch(bp0 + (ks + 1) * kstride, 0, 1);
      __builtin_prefetch(bp1 + (ks + 1) * kstride, 0, 1);
    }
    acc0 = __builtin_amdgcn_wmma_f32_16x16x32_f16(false, a.v, false, b0, (short)0, acc0, false, false);
    acc1 = __builtin_amdgcn_wmma_f32_16x16x32_f16(false, a.v, false, b1, (short)0, acc1, false, false);
  }
  const float ba = bias[col];
  const float bb = bias[col + N];
#pragma unroll
  for (int v = 0; v < 8; ++v) {
    int   row = rt * 16 + v + 8 * half;          // C/D layout: row = vgpr + 8*half
    float xa  = acc0[v] + ba;
    float xb  = acc1[v] + bb;
    outg[(size_t)row * N + col] = xa * fast_sigmoid(xb);
  }
}

// ------------------------------------------------------------- layernorm row
__global__ void __launch_bounds__(256)
k_layernorm(const float* __restrict__ x, const float* __restrict__ s, const float* __restrict__ bb,
            int N, _Float16* __restrict__ o16, int ld16,
            float* __restrict__ o32, long rs32, long cs32, long off32) {
  __shared__ float ssum[256], ssq[256];
  const int row = blockIdx.x;
  const float* xr = x + (size_t)row * N;
  float sum = 0.f, sq = 0.f;
  for (int c = threadIdx.x; c < N; c += 256) { float v = xr[c]; sum += v; sq += v * v; }
  ssum[threadIdx.x] = sum; ssq[threadIdx.x] = sq;
  __syncthreads();
  for (int st = 128; st > 0; st >>= 1) {
    if (threadIdx.x < st) {
      ssum[threadIdx.x] += ssum[threadIdx.x + st];
      ssq[threadIdx.x]  += ssq[threadIdx.x + st];
    }
    __syncthreads();
  }
  const float mu   = ssum[0] / N;
  const float var  = ssq[0] / N - mu * mu;
  const float rinv = rsqrtf(var + 1e-6f);
  for (int c = threadIdx.x; c < N; c += 256) {
    float v = (xr[c] - mu) * rinv * s[c] + bb[c];
    if (o16) o16[(size_t)row * ld16 + c] = (_Float16)v;
    if (o32) o32[(size_t)row * rs32 + (size_t)c * cs32 + off32] = v;
  }
}

// ------------------------------------- append at[:,:,-1] (f16) into A3 tail
__global__ void k_cvt_last(const float* __restrict__ at_ext, _Float16* __restrict__ A3, int t) {
  int i = blockIdx.x * blockDim.x + threadIdx.x;      // i = b*D + d
  if (i >= B_ * D_) return;
  int b = i / D_, d = i % D_;
  A3[(size_t)b * (DIN_ + D_) + DIN_ + d] = (_Float16)at_ext[(size_t)i * MEXT_ + t + 31];
}

// ------------------------------------------- per-channel NLM (nlm1 + nlm2)
__global__ void __launch_bounds__(128)
k_nlm(const float* __restrict__ st_ext,
      const _Float16* __restrict__ w1p, const float* __restrict__ b1, const float* __restrict__ T1p,
      const float* __restrict__ w2, const float* __restrict__ b2, const float* __restrict__ T2p,
      float* __restrict__ at_ext, int t) {
  const int lane = threadIdx.x & 31;
  const int wave = blockIdx.x * (blockDim.x >> 5) + (threadIdx.x >> 5);
  const int d    = wave >> 4;          // channel 0..1023
  const int rt   = wave & 15;          // 16-row tile of B
  const int half = lane >> 4;
  const int h    = lane & 15;
  const float T1 = T1p[0], T2 = T2p[0];

  AFrag a;
  const int brow = rt * 16 + h;        // A-matrix row (batch index)
  const float* strow = st_ext + ((size_t)brow * D_ + d) * MEXT_ + (t + 1);
#pragma unroll
  for (int v = 0; v < 8; ++v) {
    int kA = 2 * v + 16 * half;
    a.h[2 * v]     = (_Float16)strow[kA];
    a.h[2 * v + 1] = (_Float16)strow[kA + 1];
  }
  v16h f0 = *(const v16h*)(w1p + (((size_t)d * 2 + 0) * 32 + lane) * 16);
  v16h f1 = *(const v16h*)(w1p + (((size_t)d * 2 + 1) * 32 + lane) * 16);
  v8f acc0 = {}, acc1 = {};
  acc0 = __builtin_amdgcn_wmma_f32_16x16x32_f16(false, a.v, false, f0, (short)0, acc0, false, false);
  acc1 = __builtin_amdgcn_wmma_f32_16x16x32_f16(false, a.v, false, f1, (short)0, acc1, false, false);

  const float ba  = b1[(size_t)d * 32 + h];
  const float bbv = b1[(size_t)d * 32 + h + 16];
  const float w20 = w2[((size_t)h * 2 + 0) * D_ + d];
  const float w21 = w2[((size_t)h * 2 + 1) * D_ + d];
  const float c20 = b2[(size_t)d * 2 + 0];
  const float c21 = b2[(size_t)d * 2 + 1];

#pragma unroll
  for (int v = 0; v < 8; ++v) {
    float xa = (acc0[v] + ba)  / T1;
    float xb = (acc1[v] + bbv) / T1;
    float g  = xa * fast_sigmoid(xb);          // glu -> x[row, h]
    float p0 = g * w20;
    float p1 = g * w21;
#pragma unroll
    for (int msk = 1; msk < 16; msk <<= 1) {   // reduce over h within each half
      p0 += __shfl_xor(p0, msk, 32);
      p1 += __shfl_xor(p1, msk, 32);
    }
    if (h == 0) {                              // lanes 0 and 16 write rows v / v+8
      float ya  = (p0 + c20) / T2;
      float yb  = (p1 + c21) / T2;
      float out = ya * fast_sigmoid(yb);
      int   row = rt * 16 + v + 8 * half;
      at_ext[((size_t)row * D_ + d) * MEXT_ + 32 + t] = out;
    }
  }
}

// ---------------------------------------------------------- final synch map
__global__ void __launch_bounds__(256)
k_synch(const float* __restrict__ at_ext, const float* __restrict__ dp, float* __restrict__ out) {
  __shared__ float S[NSY_ * M_];               // S[j][m] = at[b, D-64+j, m]
  const int b = blockIdx.y;
  for (int idx = threadIdx.x; idx < NSY_ * M_; idx += 256) {
    int j = idx >> 5, m = idx & 31;
    S[idx] = at_ext[((size_t)b * D_ + (D_ - NSY_) + j) * MEXT_ + ITERS_ + m];
  }
  __syncthreads();
  int p = blockIdx.x * 256 + threadIdx.x;
  if (p >= SYNCH_) return;
  int i = 0, acc = 0;                          // decode triu pair (i <= j)
  while (acc + (NSY_ - i) <= p) { acc += NSY_ - i; ++i; }
  int j = i + (p - acc);
  float cp = fminf(fmaxf(dp[p], 0.f), 4.f);
  float r  = __expf(-cp);                      // decay[m] = r^(31-m)
  const float* Si = &S[i * M_];
  const float* Sj = &S[j * M_];
  float w = 1.f, num = 0.f, den = 0.f;
  for (int m = M_ - 1; m >= 0; --m) {
    num += w * Si[m] * Sj[m];
    den += w;
    w *= r;
  }
  out[(size_t)b * SYNCH_ + p] = num * rsqrtf(den);
}

// ------------------------------------------------- pack columns 4..35 to out
__global__ void k_pack_out(const float* __restrict__ st_ext, const float* __restrict__ at_ext,
                           float* __restrict__ out_st, float* __restrict__ out_at) {
  int i = blockIdx.x * blockDim.x + threadIdx.x;
  if (i >= B_ * D_) return;
  const float4* s = (const float4*)(st_ext + (size_t)i * MEXT_ + ITERS_);
  const float4* a = (const float4*)(at_ext + (size_t)i * MEXT_ + ITERS_);
  float4* os = (float4*)(out_st + (size_t)i * M_);
  float4* oa = (float4*)(out_at + (size_t)i * M_);
#pragma unroll
  for (int q = 0; q < 8; ++q) { os[q] = s[q]; oa[q] = a[q]; }
}

// ---------------------------------------------------------------------------
extern "C" void kernel_launch(void* const* d_in, const int* in_sizes, int n_in,
                              void* d_out, int out_size, void* d_ws, size_t ws_size,
                              hipStream_t stream) {
  (void)in_sizes; (void)n_in; (void)out_size; (void)ws_size;

  const float*         obs    = (const float*)d_in[0];
  const unsigned char* dones  = (const unsigned char*)d_in[1];   // jnp.bool_ -> 1 byte
  const float*         st_in  = (const float*)d_in[3];
  const float*         at_in  = (const float*)d_in[4];
  const float*         st0    = (const float*)d_in[5];
  const float*         at0    = (const float*)d_in[6];
  const float*         bb_w1  = (const float*)d_in[7];
  const float*         bb_b1  = (const float*)d_in[8];
  const float*         ln1s   = (const float*)d_in[9];
  const float*         ln1b   = (const float*)d_in[10];
  const float*         bb_w2  = (const float*)d_in[11];
  const float*         bb_b2  = (const float*)d_in[12];
  const float*         ln2s   = (const float*)d_in[13];
  const float*         ln2b   = (const float*)d_in[14];
  const float*         syn_w1 = (const float*)d_in[15];
  const float*         syn_b1 = (const float*)d_in[16];
  const float*         sl1s   = (const float*)d_in[17];
  const float*         sl1b   = (const float*)d_in[18];
  const float*         syn_w2 = (const float*)d_in[19];
  const float*         syn_b2 = (const float*)d_in[20];
  const float*         sl2s   = (const float*)d_in[21];
  const float*         sl2b   = (const float*)d_in[22];
  const float*         nlm1w  = (const float*)d_in[23];
  const float*         nlm1b  = (const float*)d_in[24];
  const float*         nlm1T  = (const float*)d_in[25];
  const float*         nlm2w  = (const float*)d_in[26];
  const float*         nlm2b  = (const float*)d_in[27];
  const float*         nlm2T  = (const float*)d_in[28];
  const float*         decayp = (const float*)d_in[29];

  // --- workspace carve-up (~92 MB) ---
  char*  ws  = (char*)d_ws;
  size_t off = 0;
  auto alloc = [&](size_t bytes) -> void* {
    void* p = ws + off;
    off = (off + bytes + 255) & ~((size_t)255);
    return p;
  };
  float*    st_ext  = (float*)alloc((size_t)B_ * D_ * MEXT_ * 4);
  float*    at_ext  = (float*)alloc((size_t)B_ * D_ * MEXT_ * 4);
  float*    glu_tmp = (float*)alloc((size_t)B_ * D_ * 4);                 // max 256x1024
  _Float16* w1p     = (_Float16*)alloc((size_t)OBS_ * 2 * DIN_ * 2);      // packed 256x1024
  _Float16* w2p     = (_Float16*)alloc((size_t)DIN_ * 2 * DIN_ * 2);      // packed 512x1024
  _Float16* s1p     = (_Float16*)alloc((size_t)(DIN_ + D_) * 2 * D_ * 2); // packed 1536x2048
  _Float16* s2p     = (_Float16*)alloc((size_t)D_ * 2 * D_ * 2);          // packed 1024x2048
  _Float16* n1p     = (_Float16*)alloc((size_t)D_ * 2 * 512 * 2);         // packed nlm1_w
  _Float16* obs_h   = (_Float16*)alloc((size_t)B_ * OBS_ * 2);
  _Float16* A2      = (_Float16*)alloc((size_t)B_ * DIN_ * 2);
  _Float16* A3      = (_Float16*)alloc((size_t)B_ * (DIN_ + D_) * 2);
  _Float16* A4      = (_Float16*)alloc((size_t)B_ * D_ * 2);

  const int T = 256;
  k_pack_w<<<(OBS_ * 2 * DIN_) / T, T, 0, stream>>>(bb_w1, w1p, OBS_, 2 * DIN_);
  k_pack_w<<<(DIN_ * 2 * DIN_) / T, T, 0, stream>>>(bb_w2, w2p, DIN_, 2 * DIN_);
  k_pack_w<<<((DIN_ + D_) * 2 * D_) / T, T, 0, stream>>>(syn_w1, s1p, DIN_ + D_, 2 * D_);
  k_pack_w<<<(D_ * 2 * D_) / T, T, 0, stream>>>(syn_w2, s2p, D_, 2 * D_);
  k_pack_nlm<<<(D_ * 2 * 512) / T, T, 0, stream>>>(nlm1w, n1p);
  k_cvt_f16<<<(B_ * OBS_) / T, T, 0, stream>>>(obs, obs_h, B_ * OBS_);
  k_init_ext<<<(B_ * D_) / T, T, 0, stream>>>(st_in, at_in, st0, at0, dones, st_ext, at_ext);

  // backbone: f = ln(glu(obs@w1+b1)) ; f = ln(glu(f@w2+b2)) -> A3[:, :512] (f16)
  k_gemm_glu<<<128, 128, 0, stream>>>(obs_h, OBS_, w1p, bb_b1, glu_tmp, OBS_, DIN_);
  k_layernorm<<<B_, 256, 0, stream>>>(glu_tmp, ln1s, ln1b, DIN_, A2, DIN_, nullptr, 0, 0, 0);
  k_gemm_glu<<<128, 128, 0, stream>>>(A2, DIN_, w2p, bb_b2, glu_tmp, DIN_, DIN_);
  k_layernorm<<<B_, 256, 0, stream>>>(glu_tmp, ln2s, ln2b, DIN_, A3, DIN_ + D_, nullptr, 0, 0, 0);

  for (int t = 0; t < ITERS_; ++t) {
    k_cvt_last<<<(B_ * D_) / T, T, 0, stream>>>(at_ext, A3, t);
    k_gemm_glu<<<256, 128, 0, stream>>>(A3, DIN_ + D_, s1p, syn_b1, glu_tmp, DIN_ + D_, D_);
    k_layernorm<<<B_, 256, 0, stream>>>(glu_tmp, sl1s, sl1b, D_, A4, D_, nullptr, 0, 0, 0);
    k_gemm_glu<<<256, 128, 0, stream>>>(A4, D_, s2p, syn_b2, glu_tmp, D_, D_);
    // h -> st_ext[:, :, 32+t] (strided f32 append)
    k_layernorm<<<B_, 256, 0, stream>>>(glu_tmp, sl2s, sl2b, D_, nullptr, 0,
                                        st_ext, (long)D_ * MEXT_, (long)MEXT_, (long)(M_ + t));
    // NLM: appends into at_ext[:, :, 32+t]
    k_nlm<<<(D_ * 16) / 4, 128, 0, stream>>>(st_ext, n1p, nlm1b, nlm1T,
                                             nlm2w, nlm2b, nlm2T, at_ext, t);
  }

  float* out_st = (float*)d_out;
  float* out_at = out_st + (size_t)B_ * D_ * M_;
  float* out_sy = out_at + (size_t)B_ * D_ * M_;
  k_synch<<<dim3((SYNCH_ + 255) / 256, B_), 256, 0, stream>>>(at_ext, decayp, out_sy);
  k_pack_out<<<(B_ * D_) / T, T, 0, stream>>>(st_ext, at_ext, out_st, out_at);
}
